// DenoiseKTNet_3856880632159
// MI455X (gfx1250) — matrix operand order repaired and checked by reading, not compiled, not run on hardware
//
#include <hip/hip_runtime.h>
#include <hip/hip_bf16.h>

// ---------------------------------------------------------------------------
// DenoiseKTNet forward for MI455X (gfx1250, wave32, WMMA).
// All matmuls run on v_wmma_f32_16x16x32_bf16 (bf16 operands, f32 accum).
// Weights (and the attention V tensor) are stored pre-swizzled in WMMA
// B-fragment layout so fragments load as two global b128 ops per wave --
// no LDS gathers in the GEMM hot loop.
// ---------------------------------------------------------------------------

typedef unsigned short u16;
typedef __attribute__((ext_vector_type(16))) __bf16 v16bf;
typedef __attribute__((ext_vector_type(8)))  float  v8f;

union V16 { v16bf v; u16 u[16]; uint4 q[2]; };

__device__ __forceinline__ u16 f2bf(float f) {
    unsigned u = __float_as_uint(f);
    unsigned r = u + 0x7fffu + ((u >> 16) & 1u);   // round-to-nearest-even
    return (u16)(r >> 16);
}

__device__ __forceinline__ float redmax16(float v) {
    #pragma unroll
    for (int off = 8; off; off >>= 1) v = fmaxf(v, __shfl_xor(v, off, 16));
    return v;
}
__device__ __forceinline__ float redsum16(float v) {
    #pragma unroll
    for (int off = 8; off; off >>= 1) v += __shfl_xor(v, off, 16);
    return v;
}

// ---------------------------------------------------------------------------
// Weight convert + swizzle into WMMA B-fragment layout.
// Layout: block(kt over K/32, nc over N/16) -> 32 lanes x 16 contiguous bf16.
// Lane L = half*16 + n (n = col within chunk); element i -> K row kt*32+half*16+i.
// ---------------------------------------------------------------------------
__global__ void cvt_swz_k(const float* __restrict__ in, u16* __restrict__ out,
                          int K, int N, int total) {
    int idx = blockIdx.x * blockDim.x + threadIdx.x;
    if (idx >= total) return;
    int perLayer = K * N;
    int layer = idx / perLayer;
    int r = idx - layer * perLayer;
    int i  = r & 15;
    int L2 = (r >> 4) & 31;
    int blk = r >> 9;
    int nT = N >> 4;
    int nc = blk % nT;
    int kt = blk / nT;
    int k = kt * 32 + ((L2 >> 4) << 4) + i;
    int n = nc * 16 + (L2 & 15);
    out[idx] = f2bf(in[(size_t)layer * perLayer + (size_t)k * N + n]);
}

__global__ void copy_f32_k(const float* __restrict__ in, float* __restrict__ out, int n) {
    int i = blockIdx.x * blockDim.x + threadIdx.x;
    if (i < n) out[i] = in[i];
}

// ---------------------------------------------------------------------------
// x = q + pe ; y = qa + pe  (pe computed analytically)
// ---------------------------------------------------------------------------
__global__ void pos_embed_k(const float* __restrict__ q, const float* __restrict__ qa,
                            float* __restrict__ Xf, u16* __restrict__ Xb,
                            u16* __restrict__ Yb, int total) {
    int idx = blockIdx.x * blockDim.x + threadIdx.x;
    if (idx >= total) return;
    const int D = 512, S = 512;
    int d = idx & (D - 1);
    int s = (idx >> 9) & (S - 1);
    float k = (float)(d & ~1);
    float freq = __expf(k * (-9.210340371976184f / (float)D)); // -ln(1e4)/D
    float ang  = (float)s * freq;
    float pe   = (d & 1) ? __cosf(ang) : __sinf(ang);
    float xv = q[idx] + pe;
    float yv = qa[idx] + pe;
    Xf[idx] = xv;
    Xb[idx] = f2bf(xv);
    Yb[idx] = f2bf(yv);
}

// ---------------------------------------------------------------------------
// bf16 WMMA GEMM: out[M,N] = A[M,K] @ B[K,N] + bias
// A row-major bf16; Bw pre-swizzled fragment layout (see cvt_swz_k).
// Block = 128 threads (4 waves, 2x2), block tile 64x64, wave tile 32x32,
// 4 WMMAs per K-step per wave, no LDS. Requires M%64==0, N%64==0, K%32==0.
// flags: 1=ReLU, 2=bf16 row-major out, 4=bf16 swizzled out (S=512,N=512).
// ---------------------------------------------------------------------------
__global__ __launch_bounds__(128) void gemm_bf16_k(
    const u16* __restrict__ A, const u16* __restrict__ Bw,
    const float* __restrict__ bias,
    float* __restrict__ outF, u16* __restrict__ outB,
    int M, int N, int K, int flags)
{
    const int tid  = threadIdx.x;
    const int lane = tid & 31;
    const int wave = tid >> 5;
    const int wmv = wave >> 1, wnv = wave & 1;         // 2x2 wave grid
    const int m0 = blockIdx.y * 64 + wmv * 32;
    const int n0 = blockIdx.x * 64 + wnv * 32;
    const int half = lane >> 4;
    const int l16  = lane & 15;
    const int nT = N >> 4;

    v8f acc[2][2];
    #pragma unroll
    for (int t = 0; t < 2; ++t)
        #pragma unroll
        for (int c = 0; c < 2; ++c)
            acc[t][c] = (v8f){0.f,0.f,0.f,0.f,0.f,0.f,0.f,0.f};

    for (int k0 = 0; k0 < K; k0 += 32) {
        V16 a[2], b[2];
        #pragma unroll
        for (int t = 0; t < 2; ++t) {
            const u16* ap = A + (size_t)(m0 + t * 16 + l16) * K + k0;
            a[t].q[0] = *(const uint4*)(ap + half * 16);
            a[t].q[1] = *(const uint4*)(ap + half * 16 + 8);
        }
        const int kt = k0 >> 5;
        #pragma unroll
        for (int c = 0; c < 2; ++c) {
            const u16* bp = Bw + (((size_t)kt * nT + (n0 >> 4) + c) * 32 + lane) * 16;
            b[c].q[0] = *(const uint4*)(bp);
            b[c].q[1] = *(const uint4*)(bp + 8);
        }
        #pragma unroll
        for (int t = 0; t < 2; ++t)
            #pragma unroll
            for (int c = 0; c < 2; ++c)
                acc[t][c] = __builtin_amdgcn_wmma_f32_16x16x32_bf16(
                                false, a[t].v, false, b[c].v, (short)0,
                                acc[t][c], false, false);
    }

    #pragma unroll
    for (int t = 0; t < 2; ++t) {
        #pragma unroll
        for (int c = 0; c < 2; ++c) {
            const int gcol = n0 + c * 16 + l16;
            const float bv = bias ? bias[gcol] : 0.f;
            #pragma unroll
            for (int r = 0; r < 8; ++r) {
                int grow = m0 + t * 16 + r + half * 8;
                float v = acc[t][c][r] + bv;
                if (flags & 1) v = fmaxf(v, 0.f);
                if (flags & 4) {
                    // swizzled bf16 store (V tensor: per-batch S=512 rows, N=512)
                    int bb = grow >> 9, srow = grow & 511;
                    int kt2 = srow >> 5, ii = srow & 15, h2 = (srow >> 4) & 1;
                    int L2 = h2 * 16 + (gcol & 15);
                    int nc = gcol >> 4;
                    outB[((((size_t)bb * 16 + kt2) * 32 + nc) * 32 + L2) * 16 + ii] = f2bf(v);
                } else if (flags & 2) {
                    outB[(size_t)grow * N + gcol] = f2bf(v);
                } else {
                    outF[(size_t)grow * N + gcol] = v;
                }
            }
        }
    }
}

// ---------------------------------------------------------------------------
// Flash attention, strict causal (j < i), kq_same, zero_pad on row 0.
// 1 wave per (b, h, 16-query-row tile); dk = 64; key tiles of 32.
// QK row-major bf16 (Q frags + K^T frags both contiguous);
// V pre-swizzled fragment layout -> direct b128 B-frag loads.
// ---------------------------------------------------------------------------
__global__ __launch_bounds__(32) void flash_attn_k(
    const u16* __restrict__ QK, const u16* __restrict__ Vsw,
    u16* __restrict__ O, const float* __restrict__ bfp)
{
    const int S = 512, D = 512, DK = 64;
    __shared__ u16 Ps[16 * 32];

    const int lane = threadIdx.x & 31;
    const int half = lane >> 4, l16 = lane & 15;
    const int qt = blockIdx.x;           // query tile 0..31
    const int h  = blockIdx.y;           // head 0..7
    const int b  = blockIdx.z;           // batch 0..31
    const int i0 = qt * 16;
    const float sc = (1.0f + bfp[0]) * 0.125f;   // (1+bf)/sqrt(64)

    const size_t base = (size_t)b * S * D;
    const int hc = h * DK;

    // Q fragments: contraction dims [0,32) and [32,64)
    V16 qa0, qa1;
    {
        const u16* qrow = QK + base + (size_t)(i0 + l16) * D + hc;
        qa0.q[0] = *(const uint4*)(qrow + half * 16);
        qa0.q[1] = *(const uint4*)(qrow + half * 16 + 8);
        qa1.q[0] = *(const uint4*)(qrow + 32 + half * 16);
        qa1.q[1] = *(const uint4*)(qrow + 32 + half * 16 + 8);
    }

    v8f acc[4];
    #pragma unroll
    for (int c = 0; c < 4; ++c) acc[c] = (v8f){0.f,0.f,0.f,0.f,0.f,0.f,0.f,0.f};
    float mrun[8], lrun[8];
    #pragma unroll
    for (int r = 0; r < 8; ++r) { mrun[r] = -__builtin_inff(); lrun[r] = 0.f; }

    for (int j0 = 0; j0 <= i0; j0 += 32) {
        // ---- scores: 16 x 32 tile = two 16x16 fragments ----
        v8f sf[2];
        #pragma unroll
        for (int f = 0; f < 2; ++f) {
            V16 kb0, kb1;
            const u16* krow = QK + base + (size_t)(j0 + f * 16 + l16) * D + hc;
            kb0.q[0] = *(const uint4*)(krow + half * 16);
            kb0.q[1] = *(const uint4*)(krow + half * 16 + 8);
            kb1.q[0] = *(const uint4*)(krow + 32 + half * 16);
            kb1.q[1] = *(const uint4*)(krow + 32 + half * 16 + 8);
            v8f z8 = {0.f,0.f,0.f,0.f,0.f,0.f,0.f,0.f};
            z8 = __builtin_amdgcn_wmma_f32_16x16x32_bf16(
                     false, qa0.v, false, kb0.v, (short)0, z8, false, false);
            z8 = __builtin_amdgcn_wmma_f32_16x16x32_bf16(
                     false, qa1.v, false, kb1.v, (short)0, z8, false, false);
            sf[f] = z8;
        }

        // ---- scale, mask, online softmax ----
        float p0[8], p1[8], alpha[8];
        #pragma unroll
        for (int r = 0; r < 8; ++r) {
            int grow = i0 + r + half * 8;
            int c0 = j0 + l16, c1 = j0 + 16 + l16;
            float s0 = sf[0][r] * sc; if (!(c0 < grow)) s0 = -1e32f;
            float s1 = sf[1][r] * sc; if (!(c1 < grow)) s1 = -1e32f;
            float tm = redmax16(fmaxf(s0, s1));
            float mn = fmaxf(mrun[r], tm);
            float al = __expf(mrun[r] - mn);
            float e0 = __expf(s0 - mn);
            float e1 = __expf(s1 - mn);
            float rs = redsum16(e0 + e1);
            lrun[r] = lrun[r] * al + rs;
            mrun[r] = mn;
            alpha[r] = al;
            p0[r] = e0; p1[r] = e1;
        }
        #pragma unroll
        for (int c = 0; c < 4; ++c)
            #pragma unroll
            for (int r = 0; r < 8; ++r) acc[c][r] *= alpha[r];

        // ---- relayout P (C-frag -> A-frag) through LDS ----
        __syncthreads();
        #pragma unroll
        for (int r = 0; r < 8; ++r) {
            Ps[(r + half * 8) * 32 + l16]      = f2bf(p0[r]);
            Ps[(r + half * 8) * 32 + 16 + l16] = f2bf(p1[r]);
        }
        __syncthreads();
        V16 pa;
        pa.q[0] = *(const uint4*)&Ps[l16 * 32 + half * 16];
        pa.q[1] = *(const uint4*)&Ps[l16 * 32 + half * 16 + 8];

        // ---- P @ V over 4 dk chunks (direct swizzled B-frag loads) ----
        const int ktv = j0 >> 5;
        #pragma unroll
        for (int c = 0; c < 4; ++c) {
            V16 vb;
            const u16* vp = Vsw + ((((size_t)b * 16 + ktv) * 32 + (h * 4 + c)) * 32 + lane) * 16;
            vb.q[0] = *(const uint4*)(vp);
            vb.q[1] = *(const uint4*)(vp + 8);
            acc[c] = __builtin_amdgcn_wmma_f32_16x16x32_bf16(
                         false, pa.v, false, vb.v, (short)0, acc[c], false, false);
        }
    }

    // ---- epilogue: normalize, zero_pad row 0, store bf16 ----
    #pragma unroll
    for (int r = 0; r < 8; ++r) {
        int grow = i0 + r + half * 8;
        float inv = 1.0f / lrun[r];
        #pragma unroll
        for (int c = 0; c < 4; ++c) {
            float o = acc[c][r] * inv;
            if (grow == 0) o = 0.f;
            O[base + (size_t)grow * D + hc + c * 16 + l16] = f2bf(o);
        }
    }
}

// ---------------------------------------------------------------------------
// x = LN(x + delta) * g + b ; writes f32 stream and bf16 mirror. One row/block.
// ---------------------------------------------------------------------------
__global__ __launch_bounds__(128) void add_ln_k(
    float* __restrict__ X, u16* __restrict__ Xb,
    const float* __restrict__ Dlt,
    const float* __restrict__ g, const float* __restrict__ bta)
{
    const int D = 512;
    __shared__ float red[8];
    const int row = blockIdx.x;
    const int tid = threadIdx.x;
    const float* xr = X + (size_t)row * D;
    const float* dr = Dlt + (size_t)row * D;

    float v[4];
    float s = 0.f;
    #pragma unroll
    for (int i = 0; i < 4; ++i) { v[i] = xr[tid * 4 + i] + dr[tid * 4 + i]; s += v[i]; }
    #pragma unroll
    for (int off = 16; off; off >>= 1) s += __shfl_xor(s, off, 32);
    if ((tid & 31) == 0) red[tid >> 5] = s;
    __syncthreads();
    float mu = (red[0] + red[1] + red[2] + red[3]) * (1.0f / D);

    float qs = 0.f;
    #pragma unroll
    for (int i = 0; i < 4; ++i) { float d = v[i] - mu; qs += d * d; }
    #pragma unroll
    for (int off = 16; off; off >>= 1) qs += __shfl_xor(qs, off, 32);
    if ((tid & 31) == 0) red[4 + (tid >> 5)] = qs;
    __syncthreads();
    float var = (red[4] + red[5] + red[6] + red[7]) * (1.0f / D);
    float rstd = rsqrtf(var + 1e-5f);

    #pragma unroll
    for (int i = 0; i < 4; ++i) {
        int d = tid * 4 + i;
        float nv = (v[i] - mu) * rstd * g[d] + bta[d];
        X[(size_t)row * D + d] = nv;
        Xb[(size_t)row * D + d] = f2bf(nv);
    }
}

// ---------------------------------------------------------------------------
// Host-side orchestration
// ---------------------------------------------------------------------------
extern "C" void kernel_launch(void* const* d_in, const int* in_sizes, int n_in,
                              void* d_out, int out_size, void* d_ws, size_t ws_size,
                              hipStream_t stream) {
    (void)in_sizes; (void)n_in; (void)ws_size;
    const int Bb = 32, S = 512, Dd = 512, FF = 2048, L = 4;
    const int BS = Bb * S;                     // 16384 rows
    const int NX = BS * Dd;                    // 8,388,608 activation elems

    const float* q_emb = (const float*)d_in[0];
    const float* qa_emb = (const float*)d_in[1];
    const float* bfp   = (const float*)d_in[2];
    const float* Wk = (const float*)d_in[3];
    const float* bk = (const float*)d_in[4];
    const float* Wv = (const float*)d_in[5];
    const float* bv = (const float*)d_in[6];
    const float* Wo = (const float*)d_in[7];
    const float* bo = (const float*)d_in[8];
    const float* ln1g = (const float*)d_in[9];
    const float* ln1b = (const float*)d_in[10];
    const float* W1 = (const float*)d_in[11];
    const float* b1 = (const float*)d_in[12];
    const float* W2 = (const float*)d_in[13];
    const float* b2 = (const float*)d_in[14];
    const float* ln2g = (const float*)d_in[15];
    const float* ln2b = (const float*)d_in[16];

    // workspace carve-up
    char* ws = (char*)d_ws;
    size_t off = 0;
    auto carve = [&](size_t bytes) -> char* {
        char* p = ws + off;
        off = (off + bytes + 255) & ~(size_t)255;
        return p;
    };
    float* Xf   = (float*)carve((size_t)NX * 4);
    u16*   Xb   = (u16*)  carve((size_t)NX * 2);
    u16*   Yb   = (u16*)  carve((size_t)NX * 2);
    u16*   QKb  = (u16*)  carve((size_t)NX * 2);
    u16*   Vsw  = (u16*)  carve((size_t)NX * 2);   // swizzled V
    u16*   Ab   = (u16*)  carve((size_t)NX * 2);
    u16*   Hb   = (u16*)  carve((size_t)BS * FF * 2);
    float* Fout = (float*)carve((size_t)NX * 4);
    u16* Wkb = (u16*)carve((size_t)L * Dd * Dd * 2);
    u16* Wvb = (u16*)carve((size_t)L * Dd * Dd * 2);
    u16* Wob = (u16*)carve((size_t)L * Dd * Dd * 2);
    u16* W1b = (u16*)carve((size_t)L * Dd * FF * 2);
    u16* W2b = (u16*)carve((size_t)L * FF * Dd * 2);

    // weight conversion + swizzle into B-fragment layout
    {
        int nw = L * Dd * Dd;
        cvt_swz_k<<<(nw + 255) / 256, 256, 0, stream>>>(Wk, Wkb, Dd, Dd, nw);
        cvt_swz_k<<<(nw + 255) / 256, 256, 0, stream>>>(Wv, Wvb, Dd, Dd, nw);
        cvt_swz_k<<<(nw + 255) / 256, 256, 0, stream>>>(Wo, Wob, Dd, Dd, nw);
        int nf = L * Dd * FF;
        cvt_swz_k<<<(nf + 255) / 256, 256, 0, stream>>>(W1, W1b, Dd, FF, nf);
        cvt_swz_k<<<(nf + 255) / 256, 256, 0, stream>>>(W2, W2b, FF, Dd, nf);
    }

    // positional embedding
    pos_embed_k<<<(NX + 255) / 256, 256, 0, stream>>>(q_emb, qa_emb, Xf, Xb, Yb, NX);

    const dim3 gProj(Dd / 64, BS / 64);   // N=512 GEMMs
    const dim3 gFF1(FF / 64, BS / 64);    // N=2048 GEMM
    const dim3 gAttn(S / 16, 8, Bb);

    for (int l = 0; l < L; ++l) {
        // QK projection (kq_same -> one projection used as both Q and K)
        gemm_bf16_k<<<gProj, 128, 0, stream>>>(
            Xb, Wkb + (size_t)l * Dd * Dd, bk + (size_t)l * Dd,
            nullptr, QKb, BS, Dd, Dd, /*bf16 row-major*/2);
        // V projection (from y stream) -> swizzled fragment layout
        gemm_bf16_k<<<gProj, 128, 0, stream>>>(
            Yb, Wvb + (size_t)l * Dd * Dd, bv + (size_t)l * Dd,
            nullptr, Vsw, BS, Dd, Dd, /*bf16 swizzled*/4);
        // fused causal attention
        flash_attn_k<<<gAttn, 32, 0, stream>>>(QKb, Vsw, Ab, bfp);
        // output projection (f32 out, added in LN)
        gemm_bf16_k<<<gProj, 128, 0, stream>>>(
            Ab, Wob + (size_t)l * Dd * Dd, bo + (size_t)l * Dd,
            Fout, nullptr, BS, Dd, Dd, 0);
        add_ln_k<<<BS, 128, 0, stream>>>(
            Xf, Xb, Fout, ln1g + (size_t)l * Dd, ln1b + (size_t)l * Dd);
        // FFN
        gemm_bf16_k<<<gFF1, 128, 0, stream>>>(
            Xb, W1b + (size_t)l * Dd * FF, b1 + (size_t)l * FF,
            nullptr, Hb, BS, FF, Dd, /*relu+bf16*/3);
        gemm_bf16_k<<<gProj, 128, 0, stream>>>(
            Hb, W2b + (size_t)l * FF * Dd, b2 + (size_t)l * Dd,
            Fout, nullptr, BS, Dd, FF, 0);
        add_ln_k<<<BS, 128, 0, stream>>>(
            Xf, Xb, Fout, ln2g + (size_t)l * Dd, ln2b + (size_t)l * Dd);
    }

    copy_f32_k<<<(out_size + 255) / 256, 256, 0, stream>>>(Xf, (float*)d_out, out_size);
}
